// LSTMModule_23596550324290
// MI455X (gfx1250) — matrix operand order repaired and checked by reading, not compiled, not run on hardware
//
#include <hip/hip_runtime.h>
#include <math.h>

// ---- CDNA5 (gfx1250) vector types for WMMA ----
typedef __attribute__((ext_vector_type(16))) __bf16 v16bf;
typedef __attribute__((ext_vector_type(8)))  float  v8f;

#define HH 512   // hidden
#define BB 512   // batch
#define TT 256   // time steps
#define II 128   // input size
#define N4 2048  // 4*H (packed gate columns)

union FragBF { v16bf v; uint4 q[2]; };
union BF8    { __bf16 h[8]; uint4 q; };

__device__ __forceinline__ float sigmoidf_(float x) {
    return 1.0f / (1.0f + __expf(-x));
}

// Generic-pointer -> LDS byte offset (32-bit AS3 address) for async-to-LDS ops.
__device__ __forceinline__ unsigned lds_off(const void* p) {
    return (unsigned)(unsigned long long)
        (__attribute__((address_space(3))) const void*)p;
}

// One wave-instruction async copy: each lane moves 16 bytes global->LDS.
__device__ __forceinline__ void async_b128(unsigned ldsAddr, const void* gptr) {
    asm volatile("global_load_async_to_lds_b128 %0, %1, off"
                 :: "v"(ldsAddr), "v"((unsigned long long)gptr)
                 : "memory");
}

__device__ __forceinline__ void async_wait0() {
#if __has_builtin(__builtin_amdgcn_s_wait_asynccnt)
    __builtin_amdgcn_s_wait_asynccnt(0);
#else
    asm volatile("s_wait_asynccnt 0x0" ::: "memory");
#endif
}

// ---------------------------------------------------------------------------
// One LSTM timestep for one layer:
//   G[B,2048] = concat(a0_row, a1_row) @ Wp   (bf16 WMMA, fp32 acc)
//   then fused gate epilogue: i,f,g,o -> c,h update.
// Wp layout: [K/32][2048][32] bf16, columns gate-interleaved per 32-hidden
// group: j = (h/32)*128 + gate*32 + (h%32).
// Block: 64 batch rows x 128 packed cols (= 32 hidden units, all 4 gates).
// K-loop is double-buffered: B tiles prefetched with GLOBAL_LOAD_ASYNC_TO_LDS
// (ASYNCcnt), A tiles software-pipelined through VGPRs with fp32->bf16 cvt.
// ---------------------------------------------------------------------------
__global__ __launch_bounds__(256)
void lstm_step_kernel(const float* __restrict__ a0, long a0stride, int Ka,
                      const float* __restrict__ a1,            // recurrent h_prev, stride HH
                      const __bf16* __restrict__ Wp,
                      const float* __restrict__ bias,          // packed [2048]
                      float* __restrict__ cState,
                      float* __restrict__ hOut,
                      float* __restrict__ out, long outStride,
                      int K)
{
    __shared__ alignas(16) __bf16 sA[2][64 * 32];    // 2 x 4 KB  A tiles
    __shared__ alignas(16) __bf16 sB[2][128 * 32];   // 2 x 8 KB  B tiles ([col][k])
    __shared__ float sG[64 * 128];                   // 32 KB gate pre-activations

    const int tid  = threadIdx.x;
    const int lane = tid & 31;
    const int wave = tid >> 5;
    const int mw = wave >> 2;                // 0..1 : 32-row group
    const int nw = wave & 3;                 // 0..3 : 32-col group
    const int mbase = blockIdx.y * 64;       // batch base
    const int nbase = blockIdx.x * 128;      // packed column base
    const int hbase = blockIdx.x * 32;       // hidden-unit base

    v8f acc[2][2];
    v8f zero = {};
    acc[0][0] = zero; acc[0][1] = zero; acc[1][0] = zero; acc[1][1] = zero;

    const int srow  = tid >> 2;              // 0..63  (A staging row)
    const int skcol = (tid & 3) * 8;         // 0/8/16/24 (A staging k offset)
    const int nK = K >> 5;

    // A-source pointer for a K-chunk (chunk never straddles the a0/a1 split).
    auto aSrcPtr = [&](int kbase) -> const float* {
        const bool  p0  = (kbase < Ka);
        const float* src = p0 ? a0 : a1;
        const long   rs  = p0 ? a0stride : (long)HH;
        const int    ko  = p0 ? kbase : (kbase - Ka);
        return src + (size_t)(mbase + srow) * rs + ko + skcol;
    };
    // Async-prefetch one 8 KB B tile (2 x b128 per thread).
    auto issueB = [&](int buf, int kc2) {
        const __bf16* g = Wp + (size_t)kc2 * (N4 * 32) + (size_t)nbase * 32 + tid * 8;
        unsigned l0 = lds_off(&sB[buf][tid * 8]);
        async_b128(l0,                 g);
        async_b128(l0 + 256 * 16,      g + 256 * 8);
    };
    auto storeA = [&](int buf, float4 f0, float4 f1) {
        BF8 pk;
        pk.h[0] = (__bf16)f0.x; pk.h[1] = (__bf16)f0.y;
        pk.h[2] = (__bf16)f0.z; pk.h[3] = (__bf16)f0.w;
        pk.h[4] = (__bf16)f1.x; pk.h[5] = (__bf16)f1.y;
        pk.h[6] = (__bf16)f1.z; pk.h[7] = (__bf16)f1.w;
        *(uint4*)&sA[buf][srow * 32 + skcol] = pk.q;
    };

    // ---- prologue: stage buffer 0 ----
    {
        const float* pA = aSrcPtr(0);
        float4 f0 = *(const float4*)pA;
        float4 f1 = *((const float4*)pA + 1);
        issueB(0, 0);
        storeA(0, f0, f1);
    }

    for (int kc = 0; kc < nK; ++kc) {
        const int cur = kc & 1;
        const int nxt = cur ^ 1;

        async_wait0();        // this wave's async B tile landed
        __syncthreads();      // block-wide: cur buffers ready, prev reads done

        const bool hasNext = (kc + 1 < nK);   // uniform
        float4 f0 = {0.f, 0.f, 0.f, 0.f}, f1 = {0.f, 0.f, 0.f, 0.f};
        if (hasNext) {
            issueB(nxt, kc + 1);              // overlaps with WMMAs below
            const float* pA = aSrcPtr((kc + 1) << 5);
            f0 = *(const float4*)pA;
            f1 = *((const float4*)pA + 1);
        }

        // ---- WMMA: 2x2 tiles of 16x16, K=32, on buffer `cur` ----
        const int abase = (lane >= 16) ? 8 : 0;
        const int arow0 = mw * 32 + (lane & 15);
        FragBF aF[2];
        #pragma unroll
        for (int mt = 0; mt < 2; ++mt) {
            const __bf16* ap = &sA[cur][(arow0 + mt * 16) * 32];
            aF[mt].q[0] = *(const uint4*)(ap + abase);
            aF[mt].q[1] = *(const uint4*)(ap + 16 + abase);
        }
        const int bbase = (lane >= 16) ? 8 : 0;
        const int brow0 = nw * 32 + (lane & 15);
        #pragma unroll
        for (int nt = 0; nt < 2; ++nt) {
            FragBF bF;
            const __bf16* bp = &sB[cur][(brow0 + nt * 16) * 32 + bbase];
            bF.q[0] = *(const uint4*)(bp);
            bF.q[1] = *(const uint4*)(bp + 16);
            #pragma unroll
            for (int mt = 0; mt < 2; ++mt) {
                acc[mt][nt] = __builtin_amdgcn_wmma_f32_16x16x32_bf16(
                    false, aF[mt].v, false, bF.v, (short)0, acc[mt][nt],
                    false, false);
            }
        }

        if (hasNext) storeA(nxt, f0, f1);     // consumed after next barrier
        __syncthreads();
    }

    // ---- dump accumulators to LDS (elem v -> M = v + 8*(lane>=16), N = lane%16) ----
    const int crow = (lane >= 16) ? 8 : 0;
    const int ccol = lane & 15;
    #pragma unroll
    for (int mt = 0; mt < 2; ++mt)
        #pragma unroll
        for (int nt = 0; nt < 2; ++nt)
            #pragma unroll
            for (int v = 0; v < 8; ++v)
                sG[(mw * 32 + mt * 16 + crow + v) * 128 + nw * 32 + nt * 16 + ccol]
                    = acc[mt][nt][v];
    __syncthreads();

    // ---- fused gate epilogue: 2048 (row, hidden) pairs / 256 threads ----
    #pragma unroll
    for (int i = 0; i < 8; ++i) {
        int e   = tid + i * 256;
        int row = e >> 5;
        int hh  = e & 31;
        float gi = sG[row * 128 +       hh] + bias[nbase +       hh];
        float gf = sG[row * 128 +  32 + hh] + bias[nbase +  32 + hh];
        float gg = sG[row * 128 +  64 + hh] + bias[nbase +  64 + hh];
        float go = sG[row * 128 +  96 + hh] + bias[nbase +  96 + hh];
        float ig = sigmoidf_(gi);
        float fg = sigmoidf_(gf);
        float g2 = tanhf(gg);
        float og = sigmoidf_(go);
        int b = mbase + row;
        int h = hbase + hh;
        size_t idx = (size_t)b * HH + h;
        float cOld = cState[idx];
        float cNew = fg * cOld + ig * g2;
        float hNew = og * tanhf(cNew);
        cState[idx] = cNew;
        hOut[idx]   = hNew;
        if (out) out[(size_t)b * outStride + h] = hNew;
    }
}

// ---------------------------------------------------------------------------
// Weight packing: [4H, Ka] + [4H, H] fp32 -> bf16 [K/32][2048][32],
// gate-interleaved columns: j = (h/32)*128 + gate*32 + (h%32).
// ---------------------------------------------------------------------------
__global__ void pack_weights_kernel(const float* __restrict__ Wih,
                                    const float* __restrict__ Whh,
                                    int Ka, int K, __bf16* __restrict__ Wp)
{
    int idx = blockIdx.x * 256 + threadIdx.x;
    if (idx >= K * N4) return;
    int k  = idx >> 11;
    int jp = idx & (N4 - 1);
    int hb   = jp >> 7;
    int rem  = jp & 127;
    int gate = rem >> 5;
    int hh   = rem & 31;
    int h = hb * 32 + hh;
    int r = gate * HH + h;                       // original row (gate-major)
    float v = (k < Ka) ? Wih[(size_t)r * Ka + k]
                       : Whh[(size_t)r * HH + (k - Ka)];
    Wp[(size_t)(k >> 5) * (N4 * 32) + (size_t)jp * 32 + (k & 31)] = (__bf16)v;
}

__global__ void pack_bias_kernel(const float* __restrict__ bih,
                                 const float* __restrict__ bhh,
                                 float* __restrict__ bp)
{
    int jp = blockIdx.x * 256 + threadIdx.x;
    if (jp >= N4) return;
    int hb = jp >> 7; int rem = jp & 127; int gate = rem >> 5; int hh = rem & 31;
    int r = gate * HH + hb * 32 + hh;
    bp[jp] = bih[r] + bhh[r];
}

__global__ void zero_kernel(float* __restrict__ p, int n)
{
    int i = blockIdx.x * 256 + threadIdx.x;
    if (i < n) p[i] = 0.0f;
}

// ---------------------------------------------------------------------------
extern "C" void kernel_launch(void* const* d_in, const int* in_sizes, int n_in,
                              void* d_out, int out_size, void* d_ws, size_t ws_size,
                              hipStream_t stream)
{
    (void)in_sizes; (void)n_in; (void)out_size; (void)ws_size;
    const float* x    = (const float*)d_in[0];
    const float* Wih0 = (const float*)d_in[1];
    const float* Whh0 = (const float*)d_in[2];
    const float* bih0 = (const float*)d_in[3];
    const float* bhh0 = (const float*)d_in[4];
    const float* Wih1 = (const float*)d_in[5];
    const float* Whh1 = (const float*)d_in[6];
    const float* bih1 = (const float*)d_in[7];
    const float* bhh1 = (const float*)d_in[8];
    float* out = (float*)d_out;

    const int K0 = II + HH;   // 640
    const int K1 = HH + HH;   // 1024

    char* ws = (char*)d_ws;
    size_t off = 0;
    auto wsAlloc = [&](size_t bytes) -> void* {
        void* p = ws + off;
        off = (off + bytes + 255) & ~(size_t)255;
        return p;
    };
    __bf16* Wp0 = (__bf16*)wsAlloc((size_t)K0 * N4 * sizeof(__bf16)); // 2.5 MB
    __bf16* Wp1 = (__bf16*)wsAlloc((size_t)K1 * N4 * sizeof(__bf16)); // 4.0 MB
    float*  bp0 = (float*)wsAlloc(N4 * sizeof(float));
    float*  bp1 = (float*)wsAlloc(N4 * sizeof(float));
    const size_t stateBytes = (size_t)BB * HH * sizeof(float);        // 1 MB each
    float* h1A = (float*)wsAlloc(stateBytes);
    float* h1B = (float*)wsAlloc(stateBytes);
    float* h2A = (float*)wsAlloc(stateBytes);
    float* h2B = (float*)wsAlloc(stateBytes);
    float* c1  = (float*)wsAlloc(stateBytes);
    float* c2  = (float*)wsAlloc(stateBytes);

    // --- prep (cheap, done every call for determinism) ---
    pack_weights_kernel<<<(K0 * N4 + 255) / 256, 256, 0, stream>>>(Wih0, Whh0, II, K0, Wp0);
    pack_weights_kernel<<<(K1 * N4 + 255) / 256, 256, 0, stream>>>(Wih1, Whh1, HH, K1, Wp1);
    pack_bias_kernel<<<(N4 + 255) / 256, 256, 0, stream>>>(bih0, bhh0, bp0);
    pack_bias_kernel<<<(N4 + 255) / 256, 256, 0, stream>>>(bih1, bhh1, bp1);
    const int nState = BB * HH;
    zero_kernel<<<(nState + 255) / 256, 256, 0, stream>>>(h1A, nState);
    zero_kernel<<<(nState + 255) / 256, 256, 0, stream>>>(h2A, nState);
    zero_kernel<<<(nState + 255) / 256, 256, 0, stream>>>(c1,  nState);
    zero_kernel<<<(nState + 255) / 256, 256, 0, stream>>>(c2,  nState);

    // --- sequential scan: 2 launches per timestep, h double-buffered ---
    dim3 grid(N4 / 128, BB / 64);   // (16, 8) = 128 blocks
    for (int t = 0; t < TT; ++t) {
        float* h1p = (t & 1) ? h1B : h1A;
        float* h1c = (t & 1) ? h1A : h1B;
        float* h2p = (t & 1) ? h2B : h2A;
        float* h2c = (t & 1) ? h2A : h2B;
        // layer 0: A = [x_t | h1_prev], K = 640
        lstm_step_kernel<<<grid, 256, 0, stream>>>(
            x + (size_t)t * II, (long)TT * II, II,
            h1p, Wp0, bp0, c1, h1c,
            nullptr, 0, K0);
        // layer 1: A = [h1_cur | h2_prev], K = 1024, writes d_out[:, t, :]
        lstm_step_kernel<<<grid, 256, 0, stream>>>(
            h1c, (long)HH, HH,
            h2p, Wp1, bp1, c2, h2c,
            out + (size_t)t * HH, (long)TT * HH, K1);
    }
}
